// CrossStockAttention_66133906424150
// MI455X (gfx1250) — matrix operand order repaired and checked by reading, not compiled, not run on hardware
//
#include <hip/hip_runtime.h>
#include <hip/hip_bf16.h>
#include <stdint.h>

#define BB 4
#define NN 2048
#define DD 512
#define HH 8
#define TOPKK 40
#define DHH 64

typedef __attribute__((ext_vector_type(16))) _Float16 v16h;
typedef __attribute__((ext_vector_type(8)))  float    v8f;

#define NEG_INF (-3.0e38f)

__device__ __forceinline__ int lane_id(){ return threadIdx.x & 31; }
__device__ __forceinline__ int wave_id(){ return threadIdx.x >> 5; }

// gfx1250 WMMA 16x16x32 f16 fragment coordinate helpers (wave32)
// A (16x32, MxK): row m = lane&15; VGPR v holds K pair at:
__device__ __forceinline__ int a_k(int lane, int v){
  return ((v & 3) << 1) + ((v >> 2) << 4) + ((lane >> 4) << 3);
}
// B (32x16, KxN): col n = lane&15; VGPR v holds K pair at:
__device__ __forceinline__ int b_k(int lane, int v){
  return ((v & 3) << 1) + ((v >> 2) << 4) + ((lane >> 4) << 3);
}
// D (16x16 f32): col n = lane&15; VGPR v -> row m = v + 8*(lane>=16)

__device__ __forceinline__ v8f wmma_f16(v16h a, v16h b, v8f c){
  return __builtin_amdgcn_wmma_f32_16x16x32_f16(false, a, false, b, (short)0, c,
                                                false, false);
}

__device__ __forceinline__ float wave_sum(float v){
#pragma unroll
  for (int off = 16; off > 0; off >>= 1) v += __shfl_xor(v, off, 32);
  return v;
}
__device__ __forceinline__ float wave_max(float v){
#pragma unroll
  for (int off = 16; off > 0; off >>= 1) v = fmaxf(v, __shfl_xor(v, off, 32));
  return v;
}

// ---------------- kernel 1: convert weights to f16 ----------------
__global__ void convert_w_kernel(const float* __restrict__ wq,
                                 const float* __restrict__ wo,
                                 _Float16* __restrict__ wqh,
                                 _Float16* __restrict__ woh){
  int i = blockIdx.x * blockDim.x + threadIdx.x;
  const int n1 = 3 * DD * DD;
  const int n2 = DD * DD;
  if (i < n1) wqh[i] = (_Float16)wq[i];
  if (i < n2) woh[i] = (_Float16)wo[i];
}

// ---------------- kernel 2: validity bitmask per batch ----------------
__global__ void validbits_kernel(const unsigned char* __restrict__ valid,
                                 uint32_t* __restrict__ vbits,
                                 int* __restrict__ anyv){
  int b = blockIdx.x, j = threadIdx.x;   // 64 threads
  uint32_t w = 0;
  for (int k = 0; k < 32; ++k)
    if (valid[b * NN + j * 32 + k]) w |= 1u << k;
  vbits[b * 64 + j] = w;
  __shared__ uint32_t sw[64];
  sw[j] = w;
  __syncthreads();
  if (j == 0){
    uint32_t o = 0;
    for (int t = 0; t < 64; ++t) o |= sw[t];
    anyv[b] = (o != 0);
  }
}

// ------- kernel 3: L2-normalize rows -> split f16 (hi + lo) -------
__global__ void normalize_kernel(const float* __restrict__ x,
                                 _Float16* __restrict__ xh,
                                 _Float16* __restrict__ xl){
  int row  = blockIdx.x * (blockDim.x >> 5) + wave_id();   // 8 waves/block
  int lane = lane_id();
  const float* r = x + (size_t)row * DD;
  float ss = 0.f;
#pragma unroll
  for (int i = 0; i < 16; ++i){ float v = r[lane + 32 * i]; ss += v * v; }
  ss = wave_sum(ss);
  float rn = 1.0f / fmaxf(sqrtf(ss), 1e-12f);
  _Float16* oh = xh + (size_t)row * DD;
  _Float16* ol = xl + (size_t)row * DD;
#pragma unroll
  for (int i = 0; i < 16; ++i){
    int c = lane + 32 * i;
    float v = r[c] * rn;
    _Float16 hi = (_Float16)v;
    oh[c] = hi;
    ol[c] = (_Float16)(v - (float)hi);
  }
}

// ---------------- kernel 4: QKV projection GEMM (WMMA) ----------------
// qkv[row, n] = sum_k x[row,k] * W[n,k] + bias[n];  M=8192, N=1536, K=512
__global__ void qkv_kernel(const float* __restrict__ x,
                           const _Float16* __restrict__ wqh,
                           const float* __restrict__ bias,
                           _Float16* __restrict__ qkvh){
  int bid = blockIdx.x;
  int mt = bid / 24, ng = bid % 24;
  int nt = ng * 4 + wave_id();
  int lane = lane_id();
  int mrow = mt * 16 + (lane & 15);
  int ncol = nt * 16 + (lane & 15);
  const float*    arow = x   + (size_t)mrow * DD;
  const _Float16* brow = wqh + (size_t)ncol * DD;   // B[k,n] = W[n,k], pairs contiguous
  v8f acc = {};
  for (int kb = 0; kb < DD; kb += 32){
    v16h af, bf;
#pragma unroll
    for (int v = 0; v < 8; ++v){
      int ka = kb + a_k(lane, v);
      af[2*v]   = (_Float16)arow[ka];
      af[2*v+1] = (_Float16)arow[ka + 1];
      int kk = kb + b_k(lane, v);
      bf[2*v]   = brow[kk];
      bf[2*v+1] = brow[kk + 1];
    }
    acc = wmma_f16(af, bf, acc);
  }
  float bv = bias[ncol];
#pragma unroll
  for (int v = 0; v < 8; ++v){
    int m = v + ((lane >> 4) << 3);
    int row = mt * 16 + m;
    qkvh[(size_t)row * (3 * DD) + ncol] = (_Float16)(acc[v] + bv);
  }
}

// ---- kernel 5: split-f16 cosine-sim WMMA + per-row top-40 -> bitmask ----
// sim = hi*hi + hi*lo + lo*hi  (~f32-accurate ranking for top-k)
__global__ void sim_topk_kernel(const _Float16* __restrict__ xh,
                                const _Float16* __restrict__ xl,
                                const uint32_t* __restrict__ vbits,
                                const int* __restrict__ anyv,
                                const unsigned char* __restrict__ valid,
                                uint32_t* __restrict__ amask){
  extern __shared__ char smem[];
  float* S = (float*)smem;                 // [16][2048] = 128KB of 320KB WGP LDS
  int bid = blockIdx.x;
  int b = bid >> 7, qt = bid & 127;
  int lane = lane_id(), w = wave_id();
  const _Float16* xbh = xh + (size_t)b * NN * DD;
  const _Float16* xbl = xl + (size_t)b * NN * DD;
  int mrow = qt * 16 + (lane & 15);
  const _Float16* arh = xbh + (size_t)mrow * DD;
  const _Float16* arl = xbl + (size_t)mrow * DD;
  for (int t = 0; t < 32; ++t){
    int nt = w * 32 + t;
    int ncol = nt * 16 + (lane & 15);
    const _Float16* brh = xbh + (size_t)ncol * DD;
    const _Float16* brl = xbl + (size_t)ncol * DD;
    v8f acc = {};
    for (int kb = 0; kb < DD; kb += 32){
      v16h ah, al, bh, bl;
#pragma unroll
      for (int v = 0; v < 8; ++v){
        int ka = kb + a_k(lane, v);
        ah[2*v] = arh[ka]; ah[2*v+1] = arh[ka + 1];
        al[2*v] = arl[ka]; al[2*v+1] = arl[ka + 1];
        int kk = kb + b_k(lane, v);
        bh[2*v] = brh[kk]; bh[2*v+1] = brh[kk + 1];
        bl[2*v] = brl[kk]; bl[2*v+1] = brl[kk + 1];
      }
      acc = wmma_f16(ah, bh, acc);
      acc = wmma_f16(ah, bl, acc);
      acc = wmma_f16(al, bh, acc);
    }
#pragma unroll
    for (int v = 0; v < 8; ++v){
      int m   = v + ((lane >> 4) << 3);
      int col = nt * 16 + (lane & 15);
      uint32_t vb = vbits[b * 64 + (col >> 5)];
      S[m * NN + col] = ((vb >> (col & 31)) & 1u) ? acc[v] : NEG_INF;
    }
  }
  __syncthreads();
  // top-40 per row: wave handles 4 rows; lane owns cols lane+32j (j<64)
  for (int rr = 0; rr < 4; ++rr){
    int m  = w * 4 + rr;
    int qg = qt * 16 + m;
    const float* Sr = S + (size_t)m * NN;
    unsigned long long sel = 0ull;         // per-lane selected bits (own columns)
    for (int it = 0; it < TOPKK; ++it){
      float best = NEG_INF; int bidx = 0x7fffffff;
      for (int j = 0; j < 64; ++j){
        if ((sel >> j) & 1ull) continue;
        int col = lane + (j << 5);
        float v = Sr[col];
        if (v > best || (v == best && col < bidx)){ best = v; bidx = col; }
      }
#pragma unroll
      for (int off = 16; off > 0; off >>= 1){
        float ov = __shfl_xor(best, off, 32);
        int   oi = __shfl_xor(bidx, off, 32);
        if (ov > best || (ov == best && oi < bidx)){ best = ov; bidx = oi; }
      }
      if ((bidx & 31) == lane) sel |= 1ull << (bidx >> 5);
    }
    int vq = valid[b * NN + qg];
    int av = anyv[b];
    for (int j = 0; j < 64; ++j){
      uint32_t tw = (uint32_t)__ballot((int)((sel >> j) & 1ull));
      if (lane == 0){
        uint32_t vb = vbits[b * 64 + j];
        uint32_t o;
        if (vq){
          uint32_t diag = ((qg >> 5) == j) ? (1u << (qg & 31)) : 0u;
          o = (tw | diag) & vb;            // (topk | eye&valid) & valid_keys
        } else {
          o = av ? vb : 0xFFFFFFFFu;       // invalid query: attend valid keys
        }
        amask[((size_t)b * NN + qg) * 64 + j] = o;
      }
    }
  }
}

// ---- kernel 6: attention: scores WMMA + mask + softmax + async P@V ----
__global__ void attn_kernel(const _Float16* __restrict__ qkvh,
                            const uint32_t* __restrict__ amask,
                            _Float16* __restrict__ ctxh){
  extern __shared__ char smem[];
  float* S = (float*)smem;                 // [16][2048] f32; P overlaid per-row f16
  int bid = blockIdx.x;
  int qt = bid & 127, h = (bid >> 7) & 7, b = bid >> 10;
  int lane = lane_id(), w = wave_id();
  const _Float16* qkb = qkvh + (size_t)b * NN * (3 * DD);
  int mloc = lane & 15;
  int qrow = qt * 16 + mloc;
  const _Float16* arow = qkb + (size_t)qrow * (3 * DD) + h * DHH;   // Q
  const float scale = 0.125f;              // 1/sqrt(64)
  for (int t = 0; t < 32; ++t){
    int nt = w * 32 + t;
    int ncol = nt * 16 + mloc;
    const _Float16* brow = qkb + (size_t)ncol * (3 * DD) + DD + h * DHH;  // K
    if (t + 1 < 32)
      __builtin_prefetch(brow + (size_t)16 * (3 * DD), 0, 0);  // next K tile rows
    v8f acc = {};
#pragma unroll
    for (int kb = 0; kb < DHH; kb += 32){
      v16h af, bf;
#pragma unroll
      for (int v = 0; v < 8; ++v){
        int ka = kb + a_k(lane, v);
        af[2*v]   = arow[ka];
        af[2*v+1] = arow[ka + 1];
        int kk = kb + b_k(lane, v);
        bf[2*v]   = brow[kk];
        bf[2*v+1] = brow[kk + 1];
      }
      acc = wmma_f16(af, bf, acc);
    }
#pragma unroll
    for (int v = 0; v < 8; ++v){
      int m   = v + ((lane >> 4) << 3);
      int col = nt * 16 + mloc;
      int qg  = qt * 16 + m;
      uint32_t msk = amask[((size_t)b * NN + qg) * 64 + (col >> 5)];
      S[m * NN + col] = ((msk >> (col & 31)) & 1u) ? acc[v] * scale : NEG_INF;
    }
  }
  __syncthreads();
  // softmax per row; write P as f16 in-place over the same LDS row (lockstep-safe)
  for (int rr = 0; rr < 4; ++rr){
    int m = w * 4 + rr;
    float* Sr = S + (size_t)m * NN;
    float mx = NEG_INF;
    for (int j = 0; j < 64; ++j) mx = fmaxf(mx, Sr[lane + (j << 5)]);
    mx = wave_max(mx);
    float sum = 0.f;
    for (int j = 0; j < 64; ++j){
      float s = Sr[lane + (j << 5)];
      sum += (s <= -1.0e37f) ? 0.f : __expf(s - mx);
    }
    sum = wave_sum(sum);
    float inv = (sum > 0.f) ? 1.0f / sum : 0.f;
    _Float16* Pr = (_Float16*)Sr;
    for (int j = 0; j < 64; ++j){
      int c = lane + (j << 5);
      float s = Sr[c];
      float p = (s <= -1.0e37f) ? 0.f : __expf(s - mx) * inv;
      Pr[c] = (_Float16)p;
    }
  }
  __syncthreads();
  // ctx[16,64] = P[16,2048] @ V[2048,64]; wave w -> 16 output columns.
  // V tiles (32 rows x 16 cols, f16) staged via async global->LDS copies,
  // double buffered on ASYNCcnt (async loads complete in order).
  {
    const unsigned stage_base = (unsigned)(16 * NN * sizeof(float)) + (unsigned)w * 2048u;
    const unsigned lds0 = (unsigned)(size_t)(smem + stage_base);  // low 32b = LDS offset
    v8f acc = {};
    int n = lane & 15;
    const _Float16* Pr   = (const _Float16*)(S + (size_t)(lane & 15) * NN);
    const _Float16* vcol = qkb + 2 * DD + h * DHH + w * 16;       // + row*3DD
    auto issue = [&](int buf, int kb){
      unsigned dst = lds0 + (unsigned)buf * 1024u + (unsigned)lane * 32u;
      const char* ga = (const char*)(vcol + (size_t)(kb + lane) * (3 * DD));
      asm volatile("global_load_async_to_lds_b128 %0, %1, off"
                   :: "v"(dst), "v"(ga) : "memory");
      asm volatile("global_load_async_to_lds_b128 %0, %1, off offset:16"
                   :: "v"(dst), "v"(ga) : "memory");
    };
    issue(0, 0);
    int buf = 0;
    for (int kbi = 0; kbi < 64; ++kbi){
      int kb = kbi << 5;
      if (kbi < 63){
        asm volatile("s_wait_dscnt 0x0" ::: "memory");   // WAR: prior tile reads done
        issue(buf ^ 1, kb + 32);
        asm volatile("s_wait_asynccnt 0x2" ::: "memory"); // current tile landed
      } else {
        asm volatile("s_wait_asynccnt 0x0" ::: "memory");
      }
      const _Float16* st = (const _Float16*)(smem + stage_base + (unsigned)buf * 1024u);
      v16h af, bf;
#pragma unroll
      for (int v = 0; v < 8; ++v){
        int ka = kb + a_k(lane, v);
        af[2*v]   = Pr[ka];
        af[2*v+1] = Pr[ka + 1];
        int k0 = b_k(lane, v);            // 0..31 within stage
        bf[2*v]   = st[(size_t)k0 * 16 + n];
        bf[2*v+1] = st[(size_t)(k0 + 1) * 16 + n];
      }
      acc = wmma_f16(af, bf, acc);
      buf ^= 1;
    }
#pragma unroll
    for (int v = 0; v < 8; ++v){
      int m  = v + ((lane >> 4) << 3);
      int qg = qt * 16 + m;
      ctxh[((size_t)b * NN + qg) * DD + h * DHH + w * 16 + n] = (_Float16)acc[v];
    }
  }
}

// ---- kernel 7: out-proj WMMA + residual + LayerNorm ----
__global__ void outln_kernel(const _Float16* __restrict__ ctxh,
                             const _Float16* __restrict__ woh,
                             const float* __restrict__ ob,
                             const float* __restrict__ x,
                             const float* __restrict__ g,
                             const float* __restrict__ beta,
                             float* __restrict__ out){
  __shared__ float Y[16][DD];              // 32KB
  int rt = blockIdx.x;
  int lane = lane_id(), w = wave_id();
  int mrow = rt * 16 + (lane & 15);
  const _Float16* arow = ctxh + (size_t)mrow * DD;
  for (int t = 0; t < 8; ++t){
    int nt = w * 8 + t;
    int ncol = nt * 16 + (lane & 15);
    const _Float16* brow = woh + (size_t)ncol * DD;
    v8f acc = {};
    for (int kb = 0; kb < DD; kb += 32){
      v16h af, bf;
#pragma unroll
      for (int v = 0; v < 8; ++v){
        int ka = kb + a_k(lane, v);
        af[2*v]   = arow[ka];
        af[2*v+1] = arow[ka + 1];
        int kk = kb + b_k(lane, v);
        bf[2*v]   = brow[kk];
        bf[2*v+1] = brow[kk + 1];
      }
      acc = wmma_f16(af, bf, acc);
    }
    float bv = ob[ncol];
#pragma unroll
    for (int v = 0; v < 8; ++v){
      int m = v + ((lane >> 4) << 3);
      Y[m][ncol] = acc[v] + bv;
    }
  }
  __syncthreads();
  for (int rr = 0; rr < 4; ++rr){
    int m = w * 4 + rr;
    int row = rt * 16 + m;
    const float* xr = x + (size_t)row * DD;
    float s = 0.f;
#pragma unroll
    for (int i = 0; i < 16; ++i){
      int c = lane + 32 * i;
      float y = xr[c] + Y[m][c];
      Y[m][c] = y;
      s += y;
    }
    s = wave_sum(s) * (1.0f / DD);
    float var = 0.f;
#pragma unroll
    for (int i = 0; i < 16; ++i){
      int c = lane + 32 * i;
      float d = Y[m][c] - s;
      var += d * d;
    }
    var = wave_sum(var) * (1.0f / DD);
    float rstd = rsqrtf(var + 1e-5f);
#pragma unroll
    for (int i = 0; i < 16; ++i){
      int c = lane + 32 * i;
      out[(size_t)row * DD + c] = (Y[m][c] - s) * rstd * g[c] + beta[c];
    }
  }
}

extern "C" void kernel_launch(void* const* d_in, const int* in_sizes, int n_in,
                              void* d_out, int out_size, void* d_ws, size_t ws_size,
                              hipStream_t stream){
  (void)in_sizes; (void)n_in; (void)out_size; (void)ws_size;
  const float*         x     = (const float*)d_in[0];
  const unsigned char* valid = (const unsigned char*)d_in[1];
  const float*         wq    = (const float*)d_in[2];
  const float*         bq    = (const float*)d_in[3];
  const float*         wo    = (const float*)d_in[4];
  const float*         ob    = (const float*)d_in[5];
  const float*         g     = (const float*)d_in[6];
  const float*         be    = (const float*)d_in[7];
  float*               out   = (float*)d_out;

  char* p = (char*)d_ws;
  auto take = [&](size_t bytes) -> char* {
    char* r = p;
    p += (bytes + 255) & ~(size_t)255;
    return r;
  };
  _Float16* wqh   = (_Float16*)take((size_t)3 * DD * DD * sizeof(_Float16));
  _Float16* woh   = (_Float16*)take((size_t)DD * DD * sizeof(_Float16));
  _Float16* xh    = (_Float16*)take((size_t)BB * NN * DD * sizeof(_Float16));
  _Float16* xl    = (_Float16*)take((size_t)BB * NN * DD * sizeof(_Float16));
  _Float16* qkvh  = (_Float16*)take((size_t)BB * NN * 3 * DD * sizeof(_Float16));
  _Float16* ctxh  = (_Float16*)take((size_t)BB * NN * DD * sizeof(_Float16));
  uint32_t* vbits = (uint32_t*)take((size_t)BB * 64 * sizeof(uint32_t));
  int*      anyv  = (int*)take((size_t)BB * sizeof(int));
  uint32_t* amask = (uint32_t*)take((size_t)BB * NN * 64 * sizeof(uint32_t));

  const size_t lds_sim  = (size_t)16 * NN * sizeof(float);            // 128KB
  const size_t lds_attn = lds_sim + (size_t)4 * 2 * 1024;             // +8KB V stage

  convert_w_kernel<<<(3 * DD * DD + 255) / 256, 256, 0, stream>>>(wq, wo, wqh, woh);
  validbits_kernel<<<BB, 64, 0, stream>>>(valid, vbits, anyv);
  normalize_kernel<<<BB * NN / 8, 256, 0, stream>>>(x, xh, xl);
  qkv_kernel<<<(BB * NN / 16) * (3 * DD / 64), 128, 0, stream>>>(x, wqh, bq, qkvh);
  sim_topk_kernel<<<BB * (NN / 16), 128, lds_sim, stream>>>(xh, xl, vbits, anyv, valid, amask);
  attn_kernel<<<BB * HH * (NN / 16), 128, lds_attn, stream>>>(qkvh, amask, ctxh);
  outln_kernel<<<BB * NN / 16, 128, 0, stream>>>(ctxh, woh, ob, x, g, be, out);
}
